// CrossAttention_15118284882167
// MI455X (gfx1250) — matrix operand order repaired and checked by reading, not compiled, not run on hardware
//
#include <hip/hip_runtime.h>
#include <hip/hip_bf16.h>
#include <math.h>

// ---------------------------------------------------------------------------
// Types for CDNA5 WMMA (wave32, v_wmma_f32_16x16x32_bf16)
// ---------------------------------------------------------------------------
typedef __bf16 bf16_t;
typedef __attribute__((ext_vector_type(16))) __bf16 bf16x16;
typedef __attribute__((ext_vector_type(8)))  float  f32x8;

union Frag {
    bf16x16 v;
    uint4   u[2];  // two 16B chunks per lane (16 bf16 values)
};

#define WMMA_BF16(a, b, c) \
    __builtin_amdgcn_wmma_f32_16x16x32_bf16(false, (a), false, (b), (short)0, (c), false, false)

// ---------------------------------------------------------------------------
// Small prep kernels: f32 -> bf16 convert, and transpose-convert for weights
// ---------------------------------------------------------------------------
__global__ void conv_bf16_k(const float* __restrict__ src, bf16_t* __restrict__ dst, int n) {
    int g = blockIdx.x * blockDim.x + threadIdx.x;
    if (g < n) dst[g] = (bf16_t)src[g];
}

// src: f32 [K][N] row-major  ->  dst: bf16 [N][K] (i.e. B transposed for WMMA B-operand)
__global__ void tconv_bf16_k(const float* __restrict__ src, bf16_t* __restrict__ dst,
                             int K, int N) {
    int g = blockIdx.x * blockDim.x + threadIdx.x;
    if (g >= K * N) return;
    int n = g / K, k = g % K;
    dst[g] = (bf16_t)src[k * N + n];
}

// ---------------------------------------------------------------------------
// Generic WMMA GEMM:  C[M][128] f32 = A[M][K] bf16 @ Bt[128][K]^T bf16
// Block = 256 threads = 8 waves; each wave computes one 16x16 C tile column.
// grid.x = M/16. K must be a multiple of 32.
// ---------------------------------------------------------------------------
__global__ __launch_bounds__(256) void gemm_bf16_n128_k(
    const bf16_t* __restrict__ A, const bf16_t* __restrict__ Bt,
    float* __restrict__ C, int M, int K) {
    const int wave = threadIdx.x >> 5;
    const int lane = threadIdx.x & 31;
    const int lo16 = lane & 15;
    const bool hi  = lane >= 16;
    const int mbase = blockIdx.x * 16;
    const int ncol  = wave * 16 + lo16;

    const bf16_t* arow = A  + (size_t)(mbase + lo16) * K;
    const bf16_t* brow = Bt + (size_t)ncol * K;

    f32x8 acc = {0.f,0.f,0.f,0.f,0.f,0.f,0.f,0.f};
    for (int kb = 0; kb < K; kb += 32) {
        Frag a, b;
        // A 16x32 bf16 layout: lanes<16 hold K=[0..7],[16..23]; lanes>=16 hold [8..15],[24..31]
        const bf16_t* ap = arow + kb + (hi ? 8 : 0);
        a.u[0] = *(const uint4*)(ap);
        a.u[1] = *(const uint4*)(ap + 16);
        // B 32x16 bf16 layout: lanes<16 hold K=[0..7],[16..23] of their column; lanes>=16 hold [8..15],[24..31]
        const bf16_t* bp = brow + kb + (hi ? 8 : 0);
        b.u[0] = *(const uint4*)(bp);
        b.u[1] = *(const uint4*)(bp + 16);
        acc = WMMA_BF16(a.v, b.v, acc);
    }
    // C f32 16x16: lane<16 -> col=lane, rows 0..7 ; lane>=16 -> col=lane-16, rows 8..15
    float* cp = C + (size_t)(mbase + (hi ? 8 : 0)) * 128 + ncol;
#pragma unroll
    for (int r = 0; r < 8; ++r) cp[(size_t)r * 128] = acc[r];
}

// ---------------------------------------------------------------------------
// EVNNonLinearity: per point (b,n): norm over f(=3) per channel, 128->64->128 MLP,
// skip, rescale. One block of 128 threads per point.
// ---------------------------------------------------------------------------
__global__ __launch_bounds__(128) void ev_nonlin_k(
    const float* __restrict__ X,     // [points*3][128]
    const float* __restrict__ w1, const float* __restrict__ b1,   // [128][64], [64]
    const float* __restrict__ w2, const float* __restrict__ b2,   // [64][128], [128]
    bf16_t* __restrict__ out_bf, float* __restrict__ out_f) {
    __shared__ float nrm[128];
    __shared__ float hbuf[64];
    const int j = threadIdx.x;
    const size_t base = (size_t)blockIdx.x * 3 * 128;
    const float x0 = X[base + j], x1 = X[base + 128 + j], x2 = X[base + 256 + j];
    const float n = sqrtf(x0 * x0 + x1 * x1 + x2 * x2);
    const bool  zn = (n <= 1e-6f);
    const float np = n + 1e-6f;
    nrm[j] = np;
    __syncthreads();
    if (j < 64) {
        float h = b1[j];
        for (int t = 0; t < 128; ++t) h += nrm[t] * w1[t * 64 + j];
        hbuf[j] = (h >= 0.f) ? h : 0.2f * h;   // leaky_relu, slope 0.2
    }
    __syncthreads();
    float nb = b2[j] + np;                      // skip connection
    for (int t = 0; t < 64; ++t) nb += hbuf[t] * w2[t * 128 + j];
    const float s = zn ? 1.0f : (nb / np);
    if (out_bf) {
        bf16_t* op = out_bf + base;
        op[j] = (bf16_t)(x0 * s); op[128 + j] = (bf16_t)(x1 * s); op[256 + j] = (bf16_t)(x2 * s);
    } else {
        float* op = out_f + base;
        op[j] = x0 * s; op[128 + j] = x1 * s; op[256 + j] = x2 * s;
    }
}

// ---------------------------------------------------------------------------
// EV layernorm + pack into attention layout.
// Input xh f32 [(b*2048+n)*3 + f][h*16+d]. Output Qp bf16 [hb][n][64] (c=f*16+d, pad to 64).
// sigma = sqrt( (1/16) * sum_{f,d} x^2 ), scale = 1/(sigma+eps).
// ---------------------------------------------------------------------------
__global__ void lnpack_k(const float* __restrict__ xh, bf16_t* __restrict__ Qp) {
    int g = blockIdx.x * blockDim.x + threadIdx.x;
    if (g >= 16 * 2048) return;
    int hb = g >> 11, n = g & 2047;
    int h = hb >> 1, b = hb & 1;
    const float* xp = xh + ((size_t)(b * 2048 + n) * 3) * 128 + h * 16;
    float vals[48];
    float s = 0.f;
#pragma unroll
    for (int f = 0; f < 3; ++f)
#pragma unroll
        for (int d = 0; d < 16; ++d) {
            float v = xp[f * 128 + d];
            vals[f * 16 + d] = v;
            s += v * v;
        }
    float scale = 1.0f / (sqrtf(s * (1.0f / 16.0f)) + 1e-6f);
    bf16_t* op = Qp + ((size_t)hb * 2048 + n) * 64;
#pragma unroll
    for (int c = 0; c < 48; ++c) op[c] = (bf16_t)(vals[c] * scale);
#pragma unroll
    for (int c = 48; c < 64; ++c) op[c] = (bf16_t)0.0f;   // zero pad (read by A-fragments)
}

// Pack V transposed: Vt bf16 [hb][64][2048]; Vt[hb][c][n] = v[(b,n,f)][h*16+d].
// Rows c>=48 are never read (padding output tile eliminated) -> not written.
__global__ void packv_k(const float* __restrict__ vh, bf16_t* __restrict__ Vt) {
    int g = blockIdx.x * blockDim.x + threadIdx.x;
    if (g >= 16 * 2048) return;
    int hb = g >> 11, n = g & 2047;
    int h = hb >> 1, b = hb & 1;
    const float* xp = vh + ((size_t)(b * 2048 + n) * 3) * 128 + h * 16;
    bf16_t* op = Vt + (size_t)hb * 64 * 2048 + n;
#pragma unroll
    for (int f = 0; f < 3; ++f)
#pragma unroll
        for (int d = 0; d < 16; ++d)
            op[(size_t)(f * 16 + d) * 2048] = (bf16_t)xp[f * 128 + d];
}

// ---------------------------------------------------------------------------
// Flash attention: per (hb) pair, Q[2048][64] x K[2048][64]^T, online softmax,
// P x V via Vt[0..47][2048]. grid = (2048/128, 16), block = 256 (8 waves).
// Each wave owns 16 query rows. Softmax scale = 1/sqrt(16) = 0.25.
// The P transpose through LDS is intra-wave only: CDNA5 LDS ops from one wave
// complete in order (DScnt), so s_wait_dscnt 0 + a compiler fence replaces
// block-wide barriers.
// ---------------------------------------------------------------------------
__global__ __launch_bounds__(256) void flash_attn_k(
    const bf16_t* __restrict__ Qp, const bf16_t* __restrict__ Kp,
    const bf16_t* __restrict__ Vt, bf16_t* __restrict__ resi /* [12288][128] */) {
    __shared__ bf16_t sP[8][16][32];   // per-wave P tile (16 q-rows x 32 keys)

    const int wave = threadIdx.x >> 5;
    const int lane = threadIdx.x & 31;
    const int lo16 = lane & 15;
    const bool hi  = lane >= 16;
    const int hb = blockIdx.y;
    const int h = hb >> 1, b = hb & 1;
    const int qbase = blockIdx.x * 128 + wave * 16;

    // Q A-fragments (two K-steps: dims 0..31 and 32..63), loaded once
    const bf16_t* qrow = Qp + ((size_t)hb * 2048 + qbase + lo16) * 64;
    Frag a0, a1;
    a0.u[0] = *(const uint4*)(qrow + (hi ? 8 : 0));
    a0.u[1] = *(const uint4*)(qrow + 16 + (hi ? 8 : 0));
    a1.u[0] = *(const uint4*)(qrow + 32 + (hi ? 8 : 0));
    a1.u[1] = *(const uint4*)(qrow + 48 + (hi ? 8 : 0));

    f32x8 o0 = {0.f,0.f,0.f,0.f,0.f,0.f,0.f,0.f};
    f32x8 o1 = o0, o2 = o0;
    float m[8], l[8];
#pragma unroll
    for (int r = 0; r < 8; ++r) { m[r] = -3.0e38f; l[r] = 0.f; }

    const bf16_t* kbp = Kp + (size_t)hb * 2048 * 64;
    const bf16_t* vbp = Vt + (size_t)hb * 64 * 2048;

    for (int kb = 0; kb < 2048; kb += 32) {
        // ---- S = Q @ K^T for 32 keys (two 16x16 C tiles) ----
        const bf16_t* k0 = kbp + (size_t)(kb + lo16) * 64 + (hi ? 8 : 0);
        const bf16_t* k1 = kbp + (size_t)(kb + 16 + lo16) * 64 + (hi ? 8 : 0);
        Frag b00, b01, b10, b11;
        b00.u[0] = *(const uint4*)(k0);      b00.u[1] = *(const uint4*)(k0 + 16);
        b01.u[0] = *(const uint4*)(k0 + 32); b01.u[1] = *(const uint4*)(k0 + 48);
        b10.u[0] = *(const uint4*)(k1);      b10.u[1] = *(const uint4*)(k1 + 16);
        b11.u[0] = *(const uint4*)(k1 + 32); b11.u[1] = *(const uint4*)(k1 + 48);

        // prefetch next iteration's K/V tiles into cache (global_prefetch_b8)
        if (kb + 32 < 2048) {
            __builtin_prefetch((const void*)(k0 + 32 * 64), 0, 3);
            __builtin_prefetch((const void*)(k1 + 32 * 64), 0, 3);
            __builtin_prefetch((const void*)(vbp + (size_t)lo16 * 2048 + kb + 32 + (hi ? 16 : 0)), 0, 3);
        }

        f32x8 s0 = {0.f,0.f,0.f,0.f,0.f,0.f,0.f,0.f};
        f32x8 s1 = s0;
        s0 = WMMA_BF16(a0.v, b00.v, s0);
        s0 = WMMA_BF16(a1.v, b01.v, s0);
        s1 = WMMA_BF16(a0.v, b10.v, s1);
        s1 = WMMA_BF16(a1.v, b11.v, s1);

        // ---- online softmax (row-wise across the 16-lane half owning each C column set)
        float al[8], p0[8], p1[8];
#pragma unroll
        for (int r = 0; r < 8; ++r) {
            float v0 = s0[r] * 0.25f, v1 = s1[r] * 0.25f;
            float t = fmaxf(v0, v1);
            t = fmaxf(t, __shfl_xor(t, 1, 32));
            t = fmaxf(t, __shfl_xor(t, 2, 32));
            t = fmaxf(t, __shfl_xor(t, 4, 32));
            t = fmaxf(t, __shfl_xor(t, 8, 32));
            float mn = fmaxf(m[r], t);
            float a  = __expf(m[r] - mn);
            float e0 = __expf(v0 - mn), e1 = __expf(v1 - mn);
            float su = e0 + e1;
            su += __shfl_xor(su, 1, 32);
            su += __shfl_xor(su, 2, 32);
            su += __shfl_xor(su, 4, 32);
            su += __shfl_xor(su, 8, 32);
            l[r] = l[r] * a + su;
            m[r] = mn; al[r] = a; p0[r] = e0; p1[r] = e1;
        }
#pragma unroll
        for (int r = 0; r < 8; ++r) {
            o0[r] *= al[r]; o1[r] *= al[r]; o2[r] *= al[r];
        }

        // ---- transpose P (C layout -> A layout) through this wave's LDS slice ----
        const int prow = hi ? 8 : 0;
#pragma unroll
        for (int r = 0; r < 8; ++r) {
            sP[wave][prow + r][lo16]      = (bf16_t)p0[r];
            sP[wave][prow + r][16 + lo16] = (bf16_t)p1[r];
        }
        // intra-wave producer/consumer: LDS is in-order per wave; wait DScnt and
        // fence the compiler so the reads below are not hoisted above the stores.
        asm volatile("s_wait_dscnt 0x0" ::: "memory");
        Frag pa;
        pa.u[0] = *(const uint4*)&sP[wave][lo16][hi ? 8 : 0];
        pa.u[1] = *(const uint4*)&sP[wave][lo16][16 + (hi ? 8 : 0)];
        asm volatile("" ::: "memory");   // keep next iteration's stores after these reads

        // ---- O += P @ V (3 real output-dim tiles of 16; pad tile eliminated) ----
#pragma unroll
        for (int t = 0; t < 3; ++t) {
            const bf16_t* vp = vbp + (size_t)(t * 16 + lo16) * 2048 + kb + (hi ? 8 : 0);
            Frag bv;
            bv.u[0] = *(const uint4*)(vp);
            bv.u[1] = *(const uint4*)(vp + 16);
            f32x8& o = (t == 0) ? o0 : (t == 1) ? o1 : o2;
            o = WMMA_BF16(pa.v, bv.v, o);
        }
    }

    // ---- epilogue: normalize and scatter into resi layout [ (b,n,f) ][ h*16+d ] ----
#pragma unroll
    for (int r = 0; r < 8; ++r) {
        float inv = 1.0f / l[r];
        o0[r] *= inv; o1[r] *= inv; o2[r] *= inv;
    }
    const int rbase = hi ? 8 : 0;
#pragma unroll
    for (int t = 0; t < 3; ++t) {
        f32x8& o = (t == 0) ? o0 : (t == 1) ? o1 : o2;
#pragma unroll
        for (int r = 0; r < 8; ++r) {
            int mrow = qbase + rbase + r;
            size_t idx = ((size_t)(b * 2048 + mrow) * 3 + t) * 128 + h * 16 + lo16;
            resi[idx] = (bf16_t)o[r];
        }
    }
}

// ---------------------------------------------------------------------------
// Host-side orchestration
// ---------------------------------------------------------------------------
extern "C" void kernel_launch(void* const* d_in, const int* in_sizes, int n_in,
                              void* d_out, int out_size, void* d_ws, size_t ws_size,
                              hipStream_t stream) {
    (void)in_sizes; (void)n_in; (void)out_size; (void)ws_size;
    const float* qfts     = (const float*)d_in[0];
    const float* kvfts    = (const float*)d_in[1];
    const float* w_q_in   = (const float*)d_in[2];
    const float* q_nl_w1  = (const float*)d_in[3];
    const float* q_nl_b1  = (const float*)d_in[4];
    const float* q_nl_w2  = (const float*)d_in[5];
    const float* q_nl_b2  = (const float*)d_in[6];
    const float* w_kv_in  = (const float*)d_in[7];
    const float* kv_nl_w1 = (const float*)d_in[8];
    const float* kv_nl_b1 = (const float*)d_in[9];
    const float* kv_nl_w2 = (const float*)d_in[10];
    const float* kv_nl_b2 = (const float*)d_in[11];
    const float* w_q      = (const float*)d_in[12];
    const float* w_k      = (const float*)d_in[13];
    const float* w_v      = (const float*)d_in[14];
    const float* w_out    = (const float*)d_in[15];
    const float* o_nl_w1  = (const float*)d_in[16];
    const float* o_nl_b1  = (const float*)d_in[17];
    const float* o_nl_w2  = (const float*)d_in[18];
    const float* o_nl_b2  = (const float*)d_in[19];

    const int M = 12288;          // B*N*3 rows
    const int PTS = 4096;         // B*N points

    char* ws = (char*)d_ws;
    size_t off = 0;
    auto alloc = [&](size_t bytes) -> void* {
        void* p = ws + off;
        off += (bytes + 255) & ~(size_t)255;
        return p;
    };

    bf16_t* qftsb = (bf16_t*)alloc((size_t)M * 64 * 2);
    bf16_t* kftsb = (bf16_t*)alloc((size_t)M * 64 * 2);
    bf16_t* wqinT = (bf16_t*)alloc(64 * 128 * 2);
    bf16_t* wkvinT= (bf16_t*)alloc(64 * 128 * 2);
    bf16_t* wqT   = (bf16_t*)alloc(128 * 128 * 2);
    bf16_t* wkT   = (bf16_t*)alloc(128 * 128 * 2);
    bf16_t* wvT   = (bf16_t*)alloc(128 * 128 * 2);
    bf16_t* woutT = (bf16_t*)alloc(128 * 128 * 2);
    float*  buf1  = (float*)alloc((size_t)M * 128 * 4);  // qf0 -> qh -> out0
    float*  buf2  = (float*)alloc((size_t)M * 128 * 4);  // kf0 -> kh
    float*  buf3  = (float*)alloc((size_t)M * 128 * 4);  // vh
    bf16_t* qfbf  = (bf16_t*)alloc((size_t)M * 128 * 2); // later reused as resi_bf
    bf16_t* kfbf  = (bf16_t*)alloc((size_t)M * 128 * 2);
    bf16_t* Qp    = (bf16_t*)alloc((size_t)16 * 2048 * 64 * 2);
    bf16_t* Kp    = (bf16_t*)alloc((size_t)16 * 2048 * 64 * 2);
    bf16_t* Vt    = (bf16_t*)alloc((size_t)16 * 64 * 2048 * 2);
    bf16_t* resib = qfbf;   // reuse: qfbf dead after q-head GEMM

    // --- prep: convert activations and transpose-convert weights to bf16 ---
    conv_bf16_k<<<(M * 64 + 255) / 256, 256, 0, stream>>>(qfts,  qftsb, M * 64);
    conv_bf16_k<<<(M * 64 + 255) / 256, 256, 0, stream>>>(kvfts, kftsb, M * 64);
    tconv_bf16_k<<<(64 * 128 + 255) / 256, 256, 0, stream>>>(w_q_in,  wqinT,  64, 128);
    tconv_bf16_k<<<(64 * 128 + 255) / 256, 256, 0, stream>>>(w_kv_in, wkvinT, 64, 128);
    tconv_bf16_k<<<(128 * 128 + 255) / 256, 256, 0, stream>>>(w_q,   wqT,   128, 128);
    tconv_bf16_k<<<(128 * 128 + 255) / 256, 256, 0, stream>>>(w_k,   wkT,   128, 128);
    tconv_bf16_k<<<(128 * 128 + 255) / 256, 256, 0, stream>>>(w_v,   wvT,   128, 128);
    tconv_bf16_k<<<(128 * 128 + 255) / 256, 256, 0, stream>>>(w_out, woutT, 128, 128);

    // --- q branch: input proj + EVN + head proj ---
    gemm_bf16_n128_k<<<M / 16, 256, 0, stream>>>(qftsb, wqinT, buf1, M, 64);
    ev_nonlin_k<<<PTS, 128, 0, stream>>>(buf1, q_nl_w1, q_nl_b1, q_nl_w2, q_nl_b2,
                                         qfbf, (float*)nullptr);
    gemm_bf16_n128_k<<<M / 16, 256, 0, stream>>>(qfbf, wqT, buf1, M, 128);   // qh (reuse buf1)

    // --- kv branch: input proj + EVN + k,v head proj ---
    gemm_bf16_n128_k<<<M / 16, 256, 0, stream>>>(kftsb, wkvinT, buf2, M, 64);
    ev_nonlin_k<<<PTS, 128, 0, stream>>>(buf2, kv_nl_w1, kv_nl_b1, kv_nl_w2, kv_nl_b2,
                                         kfbf, (float*)nullptr);
    gemm_bf16_n128_k<<<M / 16, 256, 0, stream>>>(kfbf, wkT, buf2, M, 128);   // kh (reuse buf2)
    gemm_bf16_n128_k<<<M / 16, 256, 0, stream>>>(kfbf, wvT, buf3, M, 128);   // vh

    // --- layernorm + pack into attention layouts ---
    lnpack_k<<<(16 * 2048 + 255) / 256, 256, 0, stream>>>(buf1, Qp);
    lnpack_k<<<(16 * 2048 + 255) / 256, 256, 0, stream>>>(buf2, Kp);
    packv_k <<<(16 * 2048 + 255) / 256, 256, 0, stream>>>(buf3, Vt);

    // --- flash attention ---
    dim3 fgrid(2048 / 128, 16);
    flash_attn_k<<<fgrid, 256, 0, stream>>>(Qp, Kp, Vt, resib);

    // --- output projection + final EVN ---
    gemm_bf16_n128_k<<<M / 16, 256, 0, stream>>>(resib, woutT, buf1, M, 128);
    ev_nonlin_k<<<PTS, 128, 0, stream>>>(buf1, o_nl_w1, o_nl_b1, o_nl_w2, o_nl_b2,
                                         (bf16_t*)nullptr, (float*)d_out);
}
